// TTransformer_21474836480534
// MI455X (gfx1250) — compile-verified, matrix operand order checked by reading, not run on hardware
//
#include <hip/hip_runtime.h>
#include <hip/hip_bf16.h>

#define HH 8
#define TTT 256
#define EE 512
#define DDIM 64
#define NBATCH 128
#define NTROWS (NBATCH * TTT)   /* 32768 */
#define FFDIM 2048
#define TNUM 288
#define LN_EPS 1e-5f
#define SCALE 0.044194173824159223f   /* 1/sqrt(512) */

#define LDA 72    /* padded stride for 64-wide bf16 tiles  */
#define LDP 264   /* padded stride for 256-wide bf16 tiles */
#define LDSS 257  /* padded stride for f32 score rows      */

typedef __attribute__((ext_vector_type(16))) __bf16 v16bf;
typedef __attribute__((ext_vector_type(8)))  float  v8f;

// ---------------------------------------------------------------- helpers
__device__ __forceinline__ unsigned short f2bf(float f) {
  unsigned int u = __float_as_uint(f);
  u += 0x7FFFu + ((u >> 16) & 1u);          // round-to-nearest-even
  return (unsigned short)(u >> 16);
}

// 16x32 bf16 fragment from a row-major tile (stride ld ushorts; ld%8==0,
// base 16B aligned).  Per ISA layout lane L holds row M=L&15; its 16
// elements are two contiguous 8-element K-runs:
//   elems 0..7  -> K =      half*8 .. half*8+7
//   elems 8..15 -> K = 16 + half*8 .. half*8+7
// => exactly two b128 LDS loads per lane.
__device__ __forceinline__ v16bf load_frag(const unsigned short* P, int ld, int lane) {
  const int m = lane & 15, half = lane >> 4;
  const unsigned short* p = P + m * ld + half * 8;
  union { uint4 u[2]; v16bf v; } c;
  c.u[0] = *(const uint4*)p;
  c.u[1] = *(const uint4*)(p + 16);
  return c.v;
}

__device__ __forceinline__ v8f wmma_bf16(v16bf a, v16bf b, v8f c) {
  return __builtin_amdgcn_wmma_f32_16x16x32_bf16(false, a, false, b,
                                                 (short)0, c, false, false);
}

// one 16x64 stripe of (64x64 in) @ (64x64 w^T-stored) -> dst (bf16)
__device__ __forceinline__ void proj64(const unsigned short* sIn,
                                       const unsigned short* sWt,
                                       unsigned short* dst, long obase,
                                       int tbase, int m0, int lane) {
  const int nl = lane & 15, hf = lane >> 4;
#pragma unroll
  for (int nt = 0; nt < 4; ++nt) {
    v8f acc = {};
#pragma unroll
    for (int ks = 0; ks < 2; ++ks) {
      v16bf a = load_frag(sIn + m0 * LDA + ks * 32, LDA, lane);
      v16bf b = load_frag(sWt + nt * 16 * LDA + ks * 32, LDA, lane);
      acc = wmma_bf16(a, b, acc);
    }
#pragma unroll
    for (int r = 0; r < 8; ++r) {
      const int row = m0 + r + hf * 8;
      dst[obase + (long)(tbase + row) * DDIM + nt * 16 + nl] = f2bf(acc[r]);
    }
  }
}

// ================================================================ kernel 1
// temb add + per-head QKV projections.  grid = (NTROWS/64, H), block = 128.
__global__ __launch_bounds__(128) void qkv_embed_kernel(
    const float* __restrict__ value, const float* __restrict__ key,
    const float* __restrict__ query, const float* __restrict__ temb,
    const float* __restrict__ Wv, const float* __restrict__ Wk,
    const float* __restrict__ Wq, const int* __restrict__ tptr,
    unsigned short* __restrict__ qb, unsigned short* __restrict__ kb,
    unsigned short* __restrict__ vb) {
  __shared__ __align__(16) unsigned short sQ[64 * LDA], sK[64 * LDA], sV[64 * LDA];
  __shared__ __align__(16) unsigned short sWq[64 * LDA], sWk[64 * LDA], sWv[64 * LDA];

  const int tid = threadIdx.x;
  const int h = blockIdx.y;
  const int r0 = blockIdx.x * 64;          // flattened N*T row base
  const int n = r0 / TTT;
  const int tbase = r0 % TTT;
  const int tshift = tptr[0];

  for (int e = tid; e < 64 * 64; e += 128) {
    const int row = e >> 6, col = e & 63;
    int ti = (tbase + row + tshift) % TNUM;
    if (ti < 0) ti += TNUM;
    const float tadd = temb[ti * EE + h * DDIM + col];
    const long g = (long)(r0 + row) * EE + h * DDIM + col;
    sQ[row * LDA + col] = f2bf(query[g] + tadd);
    sK[row * LDA + col] = f2bf(key[g] + tadd);
    sV[row * LDA + col] = f2bf(value[g]);
    // weights stored TRANSPOSED: sW[n*LDA + k] = W[k*64 + n]
    sWq[col * LDA + row] = f2bf(Wq[e]);
    sWk[col * LDA + row] = f2bf(Wk[e]);
    sWv[col * LDA + row] = f2bf(Wv[e]);
  }
  __syncthreads();

  const int wave = tid >> 5, lane = tid & 31;
  const int m0 = wave * 16;
  const long obase = (long)(n * HH + h) * TTT * DDIM;

  proj64(sQ, sWq, qb, obase, tbase, m0, lane);
  proj64(sK, sWk, kb, obase, tbase, m0, lane);
  proj64(sV, sWv, vb, obase, tbase, m0, lane);
}

// ================================================================ kernel 2
// Attention: one block per (n,h,64 query rows). Full K/V resident in LDS.
//   sK  : 256 x LDA   (row-major [T,D]; rows of K are columns of K^T)
//   sVt : 64  x LDP   (V transposed: [D,T])
//   sQ  : 64  x LDA
//   sS  : 64  x LDSS  (f32 scores)
//   sP  : 64  x LDP   (bf16 probabilities)
#define OFF_SK 0
#define OFF_SVT (OFF_SK + TTT * LDA * 2)
#define OFF_SQ (OFF_SVT + DDIM * LDP * 2)
#define OFF_SS (OFF_SQ + 64 * LDA * 2)
#define OFF_SP (OFF_SS + 64 * LDSS * 4)
#define ATTN_LDS (OFF_SP + 64 * LDP * 2)

__global__ __launch_bounds__(128) void attn_kernel(
    const unsigned short* __restrict__ qb, const unsigned short* __restrict__ kb,
    const unsigned short* __restrict__ vb, unsigned short* __restrict__ ctx) {
  extern __shared__ __align__(16) unsigned char smem[];
  unsigned short* sK  = (unsigned short*)(smem + OFF_SK);
  unsigned short* sVt = (unsigned short*)(smem + OFF_SVT);
  unsigned short* sQ  = (unsigned short*)(smem + OFF_SQ);
  float*          sS  = (float*)(smem + OFF_SS);
  unsigned short* sP  = (unsigned short*)(smem + OFF_SP);

  const int tid = threadIdx.x;
  const int t0 = blockIdx.x * 64;
  const int nh = blockIdx.y;
  const int n = nh / HH, h = nh % HH;
  const long base = (long)nh * TTT * DDIM;

  {
    const uint4* gk = (const uint4*)(kb + base);
    const uint4* gv = (const uint4*)(vb + base);
    for (int i = tid; i < TTT * DDIM / 8; i += 128) {
      const int k = i >> 3, d0 = (i & 7) * 8;
      *(uint4*)(sK + k * LDA + d0) = gk[i];
      union { uint4 u; unsigned short s[8]; } c;
      c.u = gv[i];
#pragma unroll
      for (int j = 0; j < 8; ++j) sVt[(d0 + j) * LDP + k] = c.s[j];
    }
    const uint4* gq = (const uint4*)(qb + base + (long)t0 * DDIM);
    for (int i = tid; i < 64 * DDIM / 8; i += 128) {
      const int r = i >> 3, d0 = (i & 7) * 8;
      *(uint4*)(sQ + r * LDA + d0) = gq[i];
    }
  }
  __syncthreads();

  const int wave = tid >> 5, lane = tid & 31;
  const int nl = lane & 15, hf = lane >> 4;
  const int m0 = wave * 16;

  // ---- S = Q K^T
  for (int kt = 0; kt < 16; ++kt) {
    const int n0 = kt * 16;
    v8f acc = {};
#pragma unroll
    for (int ks = 0; ks < 2; ++ks) {
      v16bf a = load_frag(sQ + m0 * LDA + ks * 32, LDA, lane);
      v16bf b = load_frag(sK + n0 * LDA + ks * 32, LDA, lane);
      acc = wmma_bf16(a, b, acc);
    }
#pragma unroll
    for (int r = 0; r < 8; ++r)
      sS[(m0 + r + hf * 8) * LDSS + n0 + nl] = acc[r];
  }
  __syncthreads();

  // ---- exact reference semantics: mask-fill 1e6, scale, clip, softmax
  if (tid < 64) {
    float* row = sS + tid * LDSS;
    const int qidx = t0 + tid;
    float mx = -1e30f;
    for (int k = 0; k < TTT; ++k) {
      float e = (k > qidx) ? 1e6f : row[k];
      float s = fminf(fmaxf(e * SCALE, -5.f), 5.f);
      row[k] = s;
      mx = fmaxf(mx, s);
    }
    float sum = 0.f;
    for (int k = 0; k < TTT; ++k) { float e = __expf(row[k] - mx); row[k] = e; sum += e; }
    const float inv = 1.f / sum;
    unsigned short* prow = sP + tid * LDP;
    for (int k = 0; k < TTT; ++k) prow[k] = f2bf(row[k] * inv);
  }
  __syncthreads();

  // ---- O = P @ V   (V stored transposed: fast frag path)
#pragma unroll
  for (int dt = 0; dt < 4; ++dt) {
    const int n0 = dt * 16;
    v8f acc = {};
#pragma unroll
    for (int ks = 0; ks < 8; ++ks) {
      v16bf a = load_frag(sP + m0 * LDP + ks * 32, LDP, lane);
      v16bf b = load_frag(sVt + n0 * LDP + ks * 32, LDP, lane);
      acc = wmma_bf16(a, b, acc);
    }
#pragma unroll
    for (int r = 0; r < 8; ++r) {
      const int row = m0 + r + hf * 8;
      ctx[(long)(n * TTT + t0 + row) * EE + h * DDIM + n0 + nl] = f2bf(acc[r]);
    }
  }
}

// ================================================================ kernel 3
// Generic tiled bf16 GEMM: C[MxN] = A_bf16[MxK] @ B_f32[KxN] (+bias)(+ReLU)
// BM=128, BN=64, BK=64; 256 threads = 8 waves, each wave a 16x64 stripe.
// Weight tile stored transposed in LDS for the fast fragment path.
template <bool RELU, bool OUTBF>
__global__ __launch_bounds__(256) void gemm_kernel(
    const unsigned short* __restrict__ A, const float* __restrict__ B,
    const float* __restrict__ bias, float* __restrict__ Cf,
    unsigned short* __restrict__ Cb, int M, int N, int K) {
  __shared__ __align__(16) unsigned short sA[128 * LDA];
  __shared__ __align__(16) unsigned short sBt[64 * LDA];
  const int tid = threadIdx.x;
  const int nb = blockIdx.x * 64;
  const int mb = blockIdx.y * 128;
  const int wave = tid >> 5, lane = tid & 31;
  const int nl = lane & 15, hf = lane >> 4;
  const int m0 = wave * 16;

  v8f acc[4] = {};
  for (int k0 = 0; k0 < K; k0 += 64) {
    for (int i = tid; i < 1024; i += 256) {  // 128x64 ushorts, uint4 rows
      const int row = i >> 3, c = (i & 7) * 8;
      *(uint4*)(sA + row * LDA + c) =
          *(const uint4*)(A + (long)(mb + row) * K + k0 + c);
    }
    for (int i = tid; i < 4096; i += 256) {  // 64x64 f32 -> bf16, transposed
      const int kk = i >> 6, nn = i & 63;
      sBt[nn * LDA + kk] = f2bf(B[(long)(k0 + kk) * N + nb + nn]);
    }
    __syncthreads();
#pragma unroll
    for (int ks = 0; ks < 2; ++ks) {
      v16bf a = load_frag(sA + m0 * LDA + ks * 32, LDA, lane);
#pragma unroll
      for (int nt = 0; nt < 4; ++nt) {
        v16bf b = load_frag(sBt + nt * 16 * LDA + ks * 32, LDA, lane);
        acc[nt] = wmma_bf16(a, b, acc[nt]);
      }
    }
    __syncthreads();
  }
#pragma unroll
  for (int nt = 0; nt < 4; ++nt) {
    const float bv = bias[nb + nt * 16 + nl];
#pragma unroll
    for (int r = 0; r < 8; ++r) {
      const int row = mb + m0 + r + hf * 8;
      float c = acc[nt][r] + bv;
      if (RELU) c = fmaxf(c, 0.f);
      const long idx = (long)row * N + nb + nt * 16 + nl;
      if (OUTBF) Cb[idx] = f2bf(c); else Cf[idx] = c;
    }
  }
}

// ================================================================ kernel 4
// Residual + LayerNorm, one wave32 per row. Optional temb add, optional bf16 copy.
__global__ __launch_bounds__(256) void ln_res_kernel(
    const float* __restrict__ a, const float* __restrict__ res,
    const float* __restrict__ temb, const int* __restrict__ tptr,
    const float* __restrict__ g, const float* __restrict__ beta,
    float* __restrict__ of, unsigned short* __restrict__ ob) {
  const int wave = threadIdx.x >> 5, lane = threadIdx.x & 31;
  const long row = (long)blockIdx.x * 8 + wave;
  const float* arow = a + row * EE;
  const float* rrow = res + row * EE;

  float v[16];
  if (temb) {
    int ti = ((int)(row % TTT) + tptr[0]) % TNUM;
    if (ti < 0) ti += TNUM;
    const float* trow = temb + (long)ti * EE;
#pragma unroll
    for (int i = 0; i < 16; ++i) {
      const int c = lane + i * 32;
      v[i] = arow[c] + rrow[c] + trow[c];
    }
  } else {
#pragma unroll
    for (int i = 0; i < 16; ++i) {
      const int c = lane + i * 32;
      v[i] = arow[c] + rrow[c];
    }
  }
  float s = 0.f;
#pragma unroll
  for (int i = 0; i < 16; ++i) s += v[i];
#pragma unroll
  for (int off = 16; off > 0; off >>= 1) s += __shfl_xor(s, off, 32);
  const float mean = s * (1.f / EE);
  float vs = 0.f;
#pragma unroll
  for (int i = 0; i < 16; ++i) { const float d = v[i] - mean; vs += d * d; }
#pragma unroll
  for (int off = 16; off > 0; off >>= 1) vs += __shfl_xor(vs, off, 32);
  const float rstd = rsqrtf(vs * (1.f / EE) + LN_EPS);
#pragma unroll
  for (int i = 0; i < 16; ++i) {
    const int c = lane + i * 32;
    const float o = (v[i] - mean) * rstd * g[c] + beta[c];
    of[row * EE + c] = o;
    if (ob) ob[row * EE + c] = f2bf(o);
  }
}

// ================================================================ driver
extern "C" void kernel_launch(void* const* d_in, const int* in_sizes, int n_in,
                              void* d_out, int out_size, void* d_ws,
                              size_t ws_size, hipStream_t stream) {
  const float* value = (const float*)d_in[0];
  const float* key_i = (const float*)d_in[1];
  const float* query = (const float*)d_in[2];
  const float* temb  = (const float*)d_in[3];
  const float* Wv = (const float*)d_in[4];
  const float* Wk = (const float*)d_in[5];
  const float* Wq = (const float*)d_in[6];
  const float* Wo = (const float*)d_in[7];
  const float* bo = (const float*)d_in[8];
  const float* g1 = (const float*)d_in[9];
  const float* beta1 = (const float*)d_in[10];
  const float* g2 = (const float*)d_in[11];
  const float* beta2 = (const float*)d_in[12];
  const float* W1 = (const float*)d_in[13];
  const float* bf1 = (const float*)d_in[14];
  const float* W2 = (const float*)d_in[15];
  const float* bf2 = (const float*)d_in[16];
  const int* tptr = (const int*)d_in[17];
  float* out = (float*)d_out;

  // ---- workspace layout (bytes); aliases are lifetime-disjoint
  unsigned char* ws = (unsigned char*)d_ws;
  unsigned short* qb   = (unsigned short*)(ws + 0);           // 32 MB bf16
  unsigned short* kb   = (unsigned short*)(ws + 33554432ull); // 32 MB bf16
  unsigned short* vb   = (unsigned short*)(ws + 67108864ull); // 32 MB bf16
  unsigned short* ctx  = (unsigned short*)(ws + 100663296ull);// 32 MB bf16
  unsigned short* hid  = (unsigned short*)(ws + 134217728ull);// 128 MB bf16
  float*          attp = (float*)(ws + 134217728ull);         // alias: 64 MB f32 (dead before hid used)
  float*          xf   = (float*)(ws + 268435456ull);         // 64 MB f32
  unsigned short* xb   = (unsigned short*)(ws + 335544320ull);// 32 MB bf16
  float*          yf   = (float*)(ws + 0);                    // alias qb/kb: 64 MB f32 (dead then)

  (void)hipFuncSetAttribute((const void*)attn_kernel,
                            hipFuncAttributeMaxDynamicSharedMemorySize,
                            ATTN_LDS);

  // 1) temb add + QKV head projections
  qkv_embed_kernel<<<dim3(NTROWS / 64, HH), 128, 0, stream>>>(
      value, key_i, query, temb, Wv, Wk, Wq, tptr, qb, kb, vb);

  // 2) attention
  attn_kernel<<<dim3(TTT / 64, NBATCH * HH), 128, ATTN_LDS, stream>>>(
      qb, kb, vb, ctx);

  // 3) output projection: attp = ctx @ Wo + bo
  gemm_kernel<false, false><<<dim3(EE / 64, NTROWS / 128), 256, 0, stream>>>(
      ctx, Wo, bo, attp, nullptr, NTROWS, EE, EE);

  // 4) x = LN(attp + query + temb)
  ln_res_kernel<<<NTROWS / 8, 256, 0, stream>>>(
      attp, query, temb, tptr, g1, beta1, xf, xb);

  // 5) hid = ReLU(x @ W1 + bf1)  (bf16 output)
  gemm_kernel<true, true><<<dim3(FFDIM / 64, NTROWS / 128), 256, 0, stream>>>(
      xb, W1, bf1, nullptr, hid, NTROWS, FFDIM, EE);

  // 6) y = hid @ W2 + bf2
  gemm_kernel<false, false><<<dim3(EE / 64, NTROWS / 128), 256, 0, stream>>>(
      hid, W2, bf2, yf, nullptr, NTROWS, EE, FFDIM);

  // 7) out = LN(y + x)
  ln_res_kernel<<<NTROWS / 8, 256, 0, stream>>>(
      yf, xf, nullptr, nullptr, g2, beta2, out, nullptr);
}
